// Network_SNN_87582973100410
// MI455X (gfx1250) — compile-verified, matrix operand order checked by reading, not run on hardware
//
#include <hip/hip_runtime.h>
#include <hip/hip_bf16.h>

// SNN: 35 timesteps, 3 fused GEMM+LIF layers per step (~1.4 TFLOP total).
// fp16 WMMA (V_WMMA_F32_16X16X32_F16): activations exact in fp16 (spikes {0,1},
// cumulative sums are integers <= 35); weights fp16, f32 accumulate.
// Weights + all state fit MI455X 192MB L2 -> matrix-unit bound, so staging uses
// the CDNA5 direct global->LDS async path (ASYNCcnt) with LDS double buffering.

typedef _Float16 half_t;
typedef __attribute__((ext_vector_type(16))) _Float16 v16h;
typedef __attribute__((ext_vector_type(8)))  _Float16 v8h;
typedef __attribute__((ext_vector_type(8)))  float    v8f;
typedef int v4i_ __attribute__((vector_size(16)));     // matches builtin param

union Frag { v16h v; v8h h[2]; };

#define BSZ   8192
#define K1    784
#define K1P   800     // padded to 25*32
#define NMID  1200
#define NP    1280    // padded to 40*32 and 10*128 (also K of layers 2/3)
#define N3    10
#define N3P   16
#define TSTEPS 35

#define LDS_STRIDE 40 // 32-half rows padded to 40 halves (bank skew)

#if defined(__AMDGCN__) && __has_builtin(__builtin_amdgcn_global_load_async_to_lds_b128)
#define HAVE_ASYNC_LDS 1
#endif

#ifdef HAVE_ASYNC_LDS
// direct global -> LDS copy, 16B per lane, tracked by ASYNCcnt.
// builtin signature (from clang diagnostic): param0 = v4i in AS1 (global src),
// then LDS dst, imm offset, imm cpol.
#define ASYNC_CP16(ldsp, gp)                                                   \
    __builtin_amdgcn_global_load_async_to_lds_b128(                            \
        (__attribute__((address_space(1))) v4i_*)(gp),                         \
        (__attribute__((address_space(3))) v4i_*)(ldsp), 0, 0)
#if __has_builtin(__builtin_amdgcn_s_wait_asynccnt)
#define WAIT_ASYNC(n) __builtin_amdgcn_s_wait_asynccnt(n)
#else
#define WAIT_ASYNC(n) asm volatile("s_wait_asynccnt %0" ::"i"(n) : "memory")
#endif
#endif

// ---------------------------------------------------------------- RNG (PCG hash)
__device__ __forceinline__ unsigned pcg_hash(unsigned v) {
    unsigned state = v * 747796405u + 2891336453u;
    unsigned word  = ((state >> ((state >> 28u) + 4u)) ^ state) * 277803737u;
    return (word >> 22u) ^ word;
}

// ---------------------------------------------------------------- spike input gen
__global__ __launch_bounds__(256)
void snn_spike_gen(const float* __restrict__ x, half_t* __restrict__ spk, int t) {
    int idx = blockIdx.x * 256 + threadIdx.x;          // over BSZ*K1P
    if (idx >= BSZ * K1P) return;
    int b = idx / K1P, j = idx - b * K1P;
    half_t s = (half_t)0.0f;
    if (j < K1) {
        unsigned h = pcg_hash(pcg_hash((unsigned)t * 2654435761u) ^ (unsigned)(b * K1 + j));
        float u = (float)(h >> 8) * (1.0f / 16777216.0f);
        s = (u * 5.0f <= x[b * K1 + j]) ? (half_t)1.0f : (half_t)0.0f;
    }
    spk[idx] = s;
}

// ---------------------------------------------------------------- weight fp32->fp16 pad
__global__ __launch_bounds__(256)
void snn_cvt_pad(const float* __restrict__ W, half_t* __restrict__ Wh,
                 int Nrows, int Kcols, int KPad, int total) {
    int idx = blockIdx.x * 256 + threadIdx.x;          // over NPad*KPad
    if (idx >= total) return;
    int n = idx / KPad, k = idx - n * KPad;
    float v = (n < Nrows && k < Kcols) ? W[n * Kcols + k] : 0.0f;
    Wh[idx] = (half_t)v;
}

// ---------------------------------------------------------------- GEMM + LIF (wide N)
// C[m,n] = sum_k A[m,k]*W[n,k]; mem += C; spk = mem>=1; mem*=(1-spk); sum += spk
// block: 256 thr = 8 waves (4x2); tile 128(M) x 128(N); wave tile 32x64;
// K staged 32 at a time, LDS double-buffered via async global->LDS copies.
__global__ __launch_bounds__(256)
void snn_gemm_lif(const half_t* __restrict__ A, const half_t* __restrict__ W,
                  float* __restrict__ mem, half_t* __restrict__ sum,
                  int Kp, int Np) {
    __shared__ half_t As[2][128 * LDS_STRIDE];
    __shared__ half_t Ws[2][128 * LDS_STRIDE];
    const int tid  = threadIdx.x;
    const int lane = tid & 31, wave = tid >> 5;
    const int g = lane >> 4, r = lane & 15;
    const int wm = wave & 3, wn = wave >> 2;           // 4(M) x 2(N) wave grid
    const int m0 = blockIdx.x * 128;
    const int n0 = blockIdx.y * 128;

    v8f acc[2][4] = {};

    // stage one 128x32 tile of A and of W into LDS buffer `buf`
    auto stage = [&](int k0, int buf) {
#pragma unroll
        for (int rep = 0; rep < 2; ++rep) {
            int i = tid + rep * 256;
            int row = i >> 2, seg = i & 3;
            const half_t* ga = &A[(size_t)(m0 + row) * Kp + k0 + seg * 8];
            const half_t* gw = &W[(size_t)(n0 + row) * Kp + k0 + seg * 8];
            half_t* la = &As[buf][row * LDS_STRIDE + seg * 8];
            half_t* lw = &Ws[buf][row * LDS_STRIDE + seg * 8];
#ifdef HAVE_ASYNC_LDS
            ASYNC_CP16(la, ga);
            ASYNC_CP16(lw, gw);
#else
            *(v8h*)la = *(const v8h*)ga;
            *(v8h*)lw = *(const v8h*)gw;
            __builtin_prefetch(ga + 32, 0, 3);
            __builtin_prefetch(gw + 32, 0, 3);
#endif
        }
    };

    const int nK = Kp >> 5;
    stage(0, 0);
    for (int ki = 0; ki < nK; ++ki) {
        const int cur = ki & 1;
        __syncthreads();                       // prior compute on buf cur^1 done
        const bool pend = (ki + 1 < nK);
        if (pend) stage((ki + 1) << 5, cur ^ 1);
#ifdef HAVE_ASYNC_LDS
        if (pend) { WAIT_ASYNC(4); } else { WAIT_ASYNC(0); }  // buf cur landed
#endif
        __syncthreads();                       // all waves' buf cur visible

        // fragments per documented 16-bit A/B layout: lane=row, halves
        // [0..7] at K = 8g.., halves [8..15] at K = 16+8g..
        Frag a[2], b[4];
#pragma unroll
        for (int im = 0; im < 2; ++im) {
            int mr = (wm * 32 + im * 16 + r) * LDS_STRIDE;
            a[im].h[0] = *(const v8h*)&As[cur][mr + g * 8];
            a[im].h[1] = *(const v8h*)&As[cur][mr + 16 + g * 8];
        }
#pragma unroll
        for (int jn = 0; jn < 4; ++jn) {
            int nr = (wn * 64 + jn * 16 + r) * LDS_STRIDE;
            b[jn].h[0] = *(const v8h*)&Ws[cur][nr + g * 8];
            b[jn].h[1] = *(const v8h*)&Ws[cur][nr + 16 + g * 8];
        }
#pragma unroll
        for (int im = 0; im < 2; ++im)
#pragma unroll
            for (int jn = 0; jn < 4; ++jn)
                acc[im][jn] = __builtin_amdgcn_wmma_f32_16x16x32_f16(
                    false, a[im].v, false, b[jn].v,
                    (short)0, acc[im][jn], false, false);
    }

    // fused LIF epilogue. C layout: lane l, vgpr v -> m = v + 8*(l>>4), n = l&15
#pragma unroll
    for (int im = 0; im < 2; ++im)
#pragma unroll
        for (int jn = 0; jn < 4; ++jn)
#pragma unroll
            for (int v = 0; v < 8; ++v) {
                int row = m0 + wm * 32 + im * 16 + v + 8 * g;
                int col = n0 + wn * 64 + jn * 16 + r;
                size_t o = (size_t)row * Np + col;
                float mval = mem[o] + acc[im][jn][v];
                float sp   = (mval >= 1.0f) ? 1.0f : 0.0f;
                mem[o] = mval * (1.0f - sp);
                sum[o] = (half_t)((float)sum[o] + sp);
            }
}

// ---------------------------------------------------------------- GEMM + LIF (N=16, layer 3)
// block: 256 thr = 8 waves, each wave owns 16 rows; W3 tile (16x32) in LDS.
__global__ __launch_bounds__(256)
void snn_gemm_lif_n16(const half_t* __restrict__ A, const half_t* __restrict__ W,
                      float* __restrict__ mem, half_t* __restrict__ sum, int Kp) {
    __shared__ half_t Ws[16 * LDS_STRIDE];
    const int tid  = threadIdx.x;
    const int lane = tid & 31, wave = tid >> 5;
    const int g = lane >> 4, r = lane & 15;
    const int m0 = blockIdx.x * 128 + wave * 16;

    v8f acc = {};
    for (int k0 = 0; k0 < Kp; k0 += 32) {
        __syncthreads();
        if (tid < 64) {
            int row = tid >> 2, seg = tid & 3;
            *(v8h*)&Ws[row * LDS_STRIDE + seg * 8] =
                *(const v8h*)&W[(size_t)row * Kp + k0 + seg * 8];
        }
        __syncthreads();
        Frag a, b;
        const half_t* ap = &A[(size_t)(m0 + r) * Kp + k0];
        a.h[0] = *(const v8h*)&ap[g * 8];
        a.h[1] = *(const v8h*)&ap[16 + g * 8];
        b.h[0] = *(const v8h*)&Ws[r * LDS_STRIDE + g * 8];
        b.h[1] = *(const v8h*)&Ws[r * LDS_STRIDE + 16 + g * 8];
        acc = __builtin_amdgcn_wmma_f32_16x16x32_f16(
            false, a.v, false, b.v, (short)0, acc, false, false);
    }
#pragma unroll
    for (int v = 0; v < 8; ++v) {
        int row = m0 + v + 8 * g;
        size_t o = (size_t)row * N3P + r;
        float mval = mem[o] + acc[v];
        float sp   = (mval >= 1.0f) ? 1.0f : 0.0f;
        mem[o] = mval * (1.0f - sp);
        sum[o] = (half_t)((float)sum[o] + sp);
    }
}

// ---------------------------------------------------------------- finalize
__global__ __launch_bounds__(256)
void snn_finalize(const half_t* __restrict__ sum3, float* __restrict__ out) {
    int idx = blockIdx.x * 256 + threadIdx.x;          // over BSZ*N3
    if (idx >= BSZ * N3) return;
    int b = idx / N3, n = idx - b * N3;
    out[idx] = (float)sum3[b * N3P + n] * (1.0f / (float)TSTEPS);
}

// ---------------------------------------------------------------- launch
static inline size_t align256(size_t v) { return (v + 255) & ~(size_t)255; }

extern "C" void kernel_launch(void* const* d_in, const int* in_sizes, int n_in,
                              void* d_out, int out_size, void* d_ws, size_t ws_size,
                              hipStream_t stream) {
    const float* x  = (const float*)d_in[0];
    const float* W1 = (const float*)d_in[1];
    const float* W2 = (const float*)d_in[2];
    const float* W3 = (const float*)d_in[3];
    float* out = (float*)d_out;

    char* ws = (char*)d_ws;
    size_t off = 0;
    auto carve = [&](size_t bytes) { char* p = ws + off; off += align256(bytes); return p; };

    half_t* W1h  = (half_t*)carve((size_t)NP * K1P * sizeof(half_t));
    half_t* W2h  = (half_t*)carve((size_t)NP * NP  * sizeof(half_t));
    half_t* W3h  = (half_t*)carve((size_t)N3P * NP * sizeof(half_t));
    half_t* spk  = (half_t*)carve((size_t)BSZ * K1P * sizeof(half_t));
    float*  mem1 = (float*) carve((size_t)BSZ * NP * sizeof(float));
    half_t* sum1 = (half_t*)carve((size_t)BSZ * NP * sizeof(half_t));
    float*  mem2 = (float*) carve((size_t)BSZ * NP * sizeof(float));
    half_t* sum2 = (half_t*)carve((size_t)BSZ * NP * sizeof(half_t));
    float*  mem3 = (float*) carve((size_t)BSZ * N3P * sizeof(float));
    half_t* sum3 = (half_t*)carve((size_t)BSZ * N3P * sizeof(half_t));
    (void)ws_size; (void)in_sizes; (void)n_in; (void)out_size;

    // zero all state every call (deterministic, harness poisons ws once)
    (void)hipMemsetAsync(mem1, 0, (size_t)BSZ * NP * sizeof(float),  stream);
    (void)hipMemsetAsync(sum1, 0, (size_t)BSZ * NP * sizeof(half_t), stream);
    (void)hipMemsetAsync(mem2, 0, (size_t)BSZ * NP * sizeof(float),  stream);
    (void)hipMemsetAsync(sum2, 0, (size_t)BSZ * NP * sizeof(half_t), stream);
    (void)hipMemsetAsync(mem3, 0, (size_t)BSZ * N3P * sizeof(float),  stream);
    (void)hipMemsetAsync(sum3, 0, (size_t)BSZ * N3P * sizeof(half_t), stream);

    // weights -> fp16, zero-padded
    {
        int t1 = NP * K1P, t2 = NP * NP, t3 = N3P * NP;
        snn_cvt_pad<<<(t1 + 255) / 256, 256, 0, stream>>>(W1, W1h, NMID, K1,   K1P, t1);
        snn_cvt_pad<<<(t2 + 255) / 256, 256, 0, stream>>>(W2, W2h, NMID, NMID, NP,  t2);
        snn_cvt_pad<<<(t3 + 255) / 256, 256, 0, stream>>>(W3, W3h, N3,   NMID, NP,  t3);
    }

    dim3 gWide(BSZ / 128, NP / 128);   // 64 x 10
    for (int t = 0; t < TSTEPS; ++t) {
        snn_spike_gen<<<(BSZ * K1P + 255) / 256, 256, 0, stream>>>(x, spk, t);
        snn_gemm_lif<<<gWide, 256, 0, stream>>>(spk,  W1h, mem1, sum1, K1P, NP);
        snn_gemm_lif<<<gWide, 256, 0, stream>>>(sum1, W2h, mem2, sum2, NP,  NP);
        snn_gemm_lif_n16<<<BSZ / 128, 256, 0, stream>>>(sum2, W3h, mem3, sum3, NP);
    }
    snn_finalize<<<(BSZ * N3 + 255) / 256, 256, 0, stream>>>(sum3, out);
}
